// GAT_G_42760694399686
// MI455X (gfx1250) — compile-verified, matrix operand order checked by reading, not run on hardware
//
#include <hip/hip_runtime.h>

typedef __attribute__((ext_vector_type(16))) __bf16 v16bf;
typedef __attribute__((ext_vector_type(8)))  float  v8f;

union FragBF16 {
  v16bf bf;
  uint4 q[2];
};

__device__ __forceinline__ unsigned short f32_to_bf16_rne(float f) {
  unsigned int u = __float_as_uint(f);
  u += 0x7fffu + ((u >> 16) & 1u);   // round-to-nearest-even
  return (unsigned short)(u >> 16);
}

// float atomic max via int/uint ordering trick (init memory to -FLT_MAX)
__device__ __forceinline__ void atomic_max_f32(float* addr, float val) {
  if (val >= 0.0f) atomicMax((int*)addr, __float_as_int(val));
  else             atomicMin((unsigned int*)addr, __float_as_uint(val));
}

// ---------------- bf16 convert / transpose ----------------
__global__ __launch_bounds__(256)
void k_convert_bf16_v4(const float* __restrict__ src, unsigned short* __restrict__ dst, int n4) {
  int i = blockIdx.x * blockDim.x + threadIdx.x;
  if (i >= n4) return;
  float4 v = ((const float4*)src)[i];
  ushort4 o;
  o.x = f32_to_bf16_rne(v.x);
  o.y = f32_to_bf16_rne(v.y);
  o.z = f32_to_bf16_rne(v.z);
  o.w = f32_to_bf16_rne(v.w);
  ((ushort4*)dst)[i] = o;
}

__global__ __launch_bounds__(256)
void k_transpose_bf16(const float* __restrict__ W, unsigned short* __restrict__ Wt, int K, int Nc) {
  int i = blockIdx.x * blockDim.x + threadIdx.x;
  if (i >= K * Nc) return;
  int k = i / Nc, n = i - k * Nc;
  Wt[(size_t)n * K + k] = f32_to_bf16_rne(W[i]);
}

// ---------------- WMMA GEMM ----------------
// C[M,Nc] f32 = A[M,K] bf16 (row major) x B[K,Nc], B supplied transposed as Bt[Nc,K].
// One wave -> one 16-row M tile x NT 16-col N tiles; A fragment reused across NT WMMAs.
template <int NT>
__global__ __launch_bounds__(128)
void k_wmma_gemm_bf16(const unsigned short* __restrict__ A,
                      const unsigned short* __restrict__ Bt,
                      float* __restrict__ C, int M, int Nc, int K) {
  const int lane  = threadIdx.x;                                   // 0..31 (wave32)
  const int mt    = blockIdx.x;
  const int ntb   = (blockIdx.y * blockDim.y + threadIdx.y) * NT;  // first n-tile
  if (mt * 16 >= M || ntb * 16 >= Nc) return;
  const int half  = lane >> 4;              // 0: lanes 0-15, 1: lanes 16-31
  const int idx16 = lane & 15;              // M row (A) / N col (B)
  const unsigned short* arow = A + (size_t)(mt * 16 + idx16) * K;
  const unsigned short* brow[NT];
  #pragma unroll
  for (int i = 0; i < NT; ++i)
    brow[i] = Bt + (size_t)((ntb + i) * 16 + idx16) * K;

  v8f acc[NT] = {};
  for (int k = 0; k < K; k += 32) {
    FragBF16 a;
    // 16-bit A fragment 16x32: lanes 0-15 K=0..7 / 16..23, lanes 16-31 K=8..15 / 24..31
    a.q[0] = *(const uint4*)(arow + k + half * 8);
    a.q[1] = *(const uint4*)(arow + k + 16 + half * 8);
    #pragma unroll
    for (int i = 0; i < NT; ++i) {
      FragBF16 b;
      b.q[0] = *(const uint4*)(brow[i] + k + half * 8);
      b.q[1] = *(const uint4*)(brow[i] + k + 16 + half * 8);
      acc[i] = __builtin_amdgcn_wmma_f32_16x16x32_bf16(false, a.bf, false, b.bf,
                                                       (short)0, acc[i], false, false);
    }
  }
  // C/D layout: VGPR r -> M=r (lanes 0-15) or M=r+8 (lanes 16-31); N = lane&15
  #pragma unroll
  for (int i = 0; i < NT; ++i) {
    float* crow = C + (size_t)(mt * 16 + 8 * half) * Nc + (size_t)(ntb + i) * 16 + idx16;
    #pragma unroll
    for (int r = 0; r < 8; ++r) crow[(size_t)r * Nc] = acc[i][r];
  }
}

// ---------------- attention coefficients ----------------
__global__ __launch_bounds__(256)
void k_attn_coeff(const float* __restrict__ h, const float* __restrict__ a_src,
                  const float* __restrict__ a_dst, float* __restrict__ as,
                  float* __restrict__ ad, int Nn, int H, int C) {
  int idx = blockIdx.x * blockDim.x + threadIdx.x;
  if (idx >= Nn * H) return;
  int n = idx / H, hh = idx - n * H;
  const float* hp = h + (size_t)n * H * C + (size_t)hh * C;
  const float* pa = a_src + hh * C;
  const float* pb = a_dst + hh * C;
  float s0 = 0.f, s1 = 0.f;
  for (int c = 0; c < C; c += 4) {
    float4 v  = *(const float4*)(hp + c);
    float4 wa = *(const float4*)(pa + c);
    float4 wb = *(const float4*)(pb + c);
    s0 += v.x * wa.x + v.y * wa.y + v.z * wa.z + v.w * wa.w;
    s1 += v.x * wb.x + v.y * wb.y + v.z * wb.z + v.w * wb.w;
  }
  as[idx] = s0; ad[idx] = s1;
}

__global__ __launch_bounds__(256)
void k_init_ms(float* __restrict__ m, float* __restrict__ s, int n) {
  int i = blockIdx.x * blockDim.x + threadIdx.x;
  if (i < n) { m[i] = -3.402823466e38f; s[i] = 0.f; }
}

// ---------------- edge softmax passes ----------------
__global__ __launch_bounds__(256)
void k_edge_logits(const int* __restrict__ ei, const float* __restrict__ as,
                   const float* __restrict__ ad, float* __restrict__ ebuf,
                   float* __restrict__ m, int E, int Et, int H) {
  int idx = blockIdx.x * blockDim.x + threadIdx.x;
  if (idx >= Et * H) return;
  int e = idx / H, hh = idx - e * H;
  int s = (e < E) ? ei[e]     : (e - E);   // self loops appended
  int d = (e < E) ? ei[E + e] : (e - E);
  float v = as[s * H + hh] + ad[d * H + hh];
  v = (v > 0.f) ? v : 0.2f * v;            // leaky_relu 0.2
  ebuf[idx] = v;
  atomic_max_f32(&m[d * H + hh], v);
}

__global__ __launch_bounds__(256)
void k_edge_exp(const int* __restrict__ ei, float* __restrict__ ebuf,
                const float* __restrict__ m, float* __restrict__ ssum,
                int E, int Et, int H) {
  int idx = blockIdx.x * blockDim.x + threadIdx.x;
  if (idx >= Et * H) return;
  int e = idx / H, hh = idx - e * H;
  int d = (e < E) ? ei[E + e] : (e - E);
  float ex = __expf(ebuf[idx] - m[d * H + hh]);
  ebuf[idx] = ex;
  atomicAdd(&ssum[d * H + hh], ex);
}

// one thread per (edge, 4-channel quad): b128 gather of h[src], 4 atomic adds into agg[dst]
__global__ __launch_bounds__(256)
void k_edge_aggregate(const int* __restrict__ ei, const float* __restrict__ hfeat,
                      const float* __restrict__ ebuf, const float* __restrict__ ssum,
                      float* __restrict__ agg, int E, int Et, int H, int C) {
  long long idx = (long long)blockIdx.x * blockDim.x + threadIdx.x;
  const int HC  = H * C;
  const int HCQ = HC >> 2;
  if (idx >= (long long)Et * HCQ) return;
  int e = (int)(idx / HCQ);
  int q = (int)(idx - (long long)e * HCQ);
  int c0 = q << 2;
  int hh = c0 / C;                          // C multiple of 4 -> quad within one head
  int s = (e < E) ? ei[e]     : (e - E);
  int d = (e < E) ? ei[E + e] : (e - E);
  float alpha = ebuf[e * H + hh] / (ssum[d * H + hh] + 1e-16f);
  float4 hv = *(const float4*)(hfeat + (size_t)s * HC + c0);
  float* ap = agg + (size_t)d * HC + c0;
  atomicAdd(ap + 0, hv.x * alpha);
  atomicAdd(ap + 1, hv.y * alpha);
  atomicAdd(ap + 2, hv.z * alpha);
  atomicAdd(ap + 3, hv.w * alpha);
}

// ---------------- head mean + bias (+ELU) ----------------
__global__ __launch_bounds__(256)
void k_head_mean(const float* __restrict__ agg, const float* __restrict__ bias,
                 float* __restrict__ out_f32, unsigned short* __restrict__ out_bf,
                 int Nn, int H, int C, int elu) {
  int idx = blockIdx.x * blockDim.x + threadIdx.x;
  if (idx >= Nn * C) return;
  int n = idx / C, c = idx - n * C;
  float sum = 0.f;
  for (int hh = 0; hh < H; ++hh) sum += agg[(size_t)n * H * C + (size_t)hh * C + c];
  float v = sum * (1.0f / H) + bias[c];
  if (elu) v = (v > 0.f) ? v : (__expf(v) - 1.f);
  if (out_f32) out_f32[idx] = v;
  if (out_bf)  out_bf[idx]  = f32_to_bf16_rne(v);
}

// ---------------- pooling + MLP ----------------
__global__ __launch_bounds__(256)
void k_pool_accum(const float* __restrict__ h, const int* __restrict__ batch,
                  float* __restrict__ pooled, float* __restrict__ counts, int Nn, int C) {
  int idx = blockIdx.x * blockDim.x + threadIdx.x;
  if (idx >= Nn * C) return;
  int n = idx / C, c = idx - n * C;
  int g = batch[n];
  atomicAdd(&pooled[(size_t)g * C + c], h[idx]);
  if (c == 0) atomicAdd(&counts[g], 1.0f);
}

__global__ __launch_bounds__(256)
void k_pool_div(float* __restrict__ pooled, const float* __restrict__ counts, int G, int C) {
  int idx = blockIdx.x * blockDim.x + threadIdx.x;
  if (idx >= G * C) return;
  float cnt = counts[idx / C];
  pooled[idx] /= (cnt < 1.f ? 1.f : cnt);
}

__global__ __launch_bounds__(256)
void k_dense(const float* __restrict__ X, const float* __restrict__ W,
             const float* __restrict__ b, float* __restrict__ Y,
             int M, int K, int Nc, int relu) {
  int idx = blockIdx.x * blockDim.x + threadIdx.x;
  if (idx >= M * Nc) return;
  int m = idx / Nc, n = idx - m * Nc;
  float acc = b[n];
  for (int k = 0; k < K; ++k) acc += X[(size_t)m * K + k] * W[(size_t)k * Nc + n];
  if (relu && acc < 0.f) acc = 0.f;
  Y[idx] = acc;
}

static inline int cdiv_ll(long long a, int b) { return (int)((a + b - 1) / b); }

extern "C" void kernel_launch(void* const* d_in, const int* in_sizes, int n_in,
                              void* d_out, int out_size, void* d_ws, size_t ws_size,
                              hipStream_t stream) {
  (void)in_sizes; (void)n_in; (void)out_size; (void)ws_size;
  const int N = 50000, E = 800000, FIN = 128, HID = 64, H1 = 4, G = 256;
  const int Et  = E + N;          // with self loops
  const int HC1 = H1 * HID;       // 256

  const float* x    = (const float*)d_in[0];
  const int*   ei   = (const int*)d_in[1];
  const int*   batch= (const int*)d_in[2];
  const float* W1   = (const float*)d_in[3];
  const float* a1s  = (const float*)d_in[4];
  const float* a1d  = (const float*)d_in[5];
  const float* b1   = (const float*)d_in[6];
  const float* W2   = (const float*)d_in[7];
  const float* a2s  = (const float*)d_in[8];
  const float* a2d  = (const float*)d_in[9];
  const float* b2   = (const float*)d_in[10];
  const float* W3   = (const float*)d_in[11];
  const float* a3s  = (const float*)d_in[12];
  const float* a3d  = (const float*)d_in[13];
  const float* b3   = (const float*)d_in[14];
  const float* fc1W = (const float*)d_in[15];
  const float* fc1b = (const float*)d_in[16];
  const float* fc2W = (const float*)d_in[17];
  const float* fc2b = (const float*)d_in[18];
  const float* fc3W = (const float*)d_in[19];
  const float* fc3b = (const float*)d_in[20];
  float* out = (float*)d_out;

  // bump allocator over d_ws (~152 MB total)
  char* wsp = (char*)d_ws;
  size_t off = 0;
  auto alloc = [&](size_t bytes) -> void* {
    void* p = wsp + off;
    off += (bytes + 255) & ~(size_t)255;
    return p;
  };

  unsigned short* xbf    = (unsigned short*)alloc((size_t)N * FIN * 2);
  unsigned short* nodebf = (unsigned short*)alloc((size_t)N * HID * 2);
  unsigned short* w1t    = (unsigned short*)alloc((size_t)HC1 * FIN * 2);
  unsigned short* w2t    = (unsigned short*)alloc((size_t)HC1 * HID * 2);
  unsigned short* w3t    = (unsigned short*)alloc((size_t)HID * HID * 2);
  float* hbuf   = (float*)alloc((size_t)N * HC1 * 4);
  float* agg    = (float*)alloc((size_t)N * HC1 * 4);
  float* asrc   = (float*)alloc((size_t)N * H1 * 4);
  float* adst   = (float*)alloc((size_t)N * H1 * 4);
  float* mbuf   = (float*)alloc((size_t)N * H1 * 4);
  float* sbuf   = (float*)alloc((size_t)N * H1 * 4);
  float* ebuf   = (float*)alloc((size_t)Et * H1 * 4);
  float* h3     = (float*)alloc((size_t)N * HID * 4);
  float* pooled = (float*)alloc((size_t)G * HID * 4);
  float* counts = (float*)alloc((size_t)G * 4);
  float* z1     = (float*)alloc((size_t)G * 64 * 4);
  float* z2     = (float*)alloc((size_t)G * 32 * 4);

  dim3 blk(256);

  // input / weight precision prep (bf16, W transposed so B loads match A striping)
  k_convert_bf16_v4<<<cdiv_ll((long long)N * FIN / 4, 256), blk, 0, stream>>>(x, xbf, N * FIN / 4);
  k_transpose_bf16<<<cdiv_ll((long long)FIN * HC1, 256), blk, 0, stream>>>(W1, w1t, FIN, HC1);
  k_transpose_bf16<<<cdiv_ll((long long)HID * HC1, 256), blk, 0, stream>>>(W2, w2t, HID, HC1);
  k_transpose_bf16<<<cdiv_ll((long long)HID * HID, 256), blk, 0, stream>>>(W3, w3t, HID, HID);

  auto run_layer = [&](const unsigned short* Abf, int K, const unsigned short* Wt,
                       int H, int C, const float* a_s, const float* a_d,
                       const float* bias, float* out_f32, unsigned short* out_bf, int elu) {
    int HC = H * C;
    dim3 gblk(32, 4);                       // 4 waves per block
    if (HC == 256) {
      dim3 ggrid(N / 16, 1);                // 4 waves x 4 n-tiles x 16 = 256 cols
      k_wmma_gemm_bf16<4><<<ggrid, gblk, 0, stream>>>(Abf, Wt, hbuf, N, HC, K);
    } else {
      dim3 ggrid(N / 16, 1);                // 4 waves x 1 n-tile x 16 = 64 cols
      k_wmma_gemm_bf16<1><<<ggrid, gblk, 0, stream>>>(Abf, Wt, hbuf, N, HC, K);
    }
    k_attn_coeff<<<cdiv_ll((long long)N * H, 256), blk, 0, stream>>>(hbuf, a_s, a_d, asrc, adst, N, H, C);
    k_init_ms  <<<cdiv_ll((long long)N * H, 256), blk, 0, stream>>>(mbuf, sbuf, N * H);
    hipMemsetAsync(agg, 0, (size_t)N * HC * 4, stream);
    k_edge_logits  <<<cdiv_ll((long long)Et * H, 256), blk, 0, stream>>>(ei, asrc, adst, ebuf, mbuf, E, Et, H);
    k_edge_exp     <<<cdiv_ll((long long)Et * H, 256), blk, 0, stream>>>(ei, ebuf, mbuf, sbuf, E, Et, H);
    k_edge_aggregate<<<cdiv_ll((long long)Et * (HC / 4), 256), blk, 0, stream>>>(ei, hbuf, ebuf, sbuf, agg, E, Et, H, C);
    k_head_mean<<<cdiv_ll((long long)N * C, 256), blk, 0, stream>>>(agg, bias, out_f32, out_bf, N, H, C, elu);
  };

  run_layer(xbf,    FIN, w1t, H1, HID, a1s, a1d, b1, nullptr, nodebf, 1);
  run_layer(nodebf, HID, w2t, H1, HID, a2s, a2d, b2, nullptr, nodebf, 1);
  run_layer(nodebf, HID, w3t, 1,  HID, a3s, a3d, b3, h3,      nullptr, 0);

  // global mean pool + MLP head
  hipMemsetAsync(pooled, 0, (size_t)G * HID * 4, stream);
  hipMemsetAsync(counts, 0, (size_t)G * 4, stream);
  k_pool_accum<<<cdiv_ll((long long)N * HID, 256), blk, 0, stream>>>(h3, batch, pooled, counts, N, HID);
  k_pool_div  <<<cdiv_ll((long long)G * HID, 256), blk, 0, stream>>>(pooled, counts, G, HID);
  k_dense<<<cdiv_ll((long long)G * 64, 256), blk, 0, stream>>>(pooled, fc1W, fc1b, z1, G, 64, 64, 1);
  k_dense<<<cdiv_ll((long long)G * 32, 256), blk, 0, stream>>>(z1, fc2W, fc2b, z2, G, 64, 32, 1);
  k_dense<<<cdiv_ll((long long)G * 1, 256),  blk, 0, stream>>>(z2, fc3W, fc3b, out, G, 32, 1, 0);
}
